// EMACodebook_25589415150073
// MI455X (gfx1250) — matrix operand order repaired and checked by reading, not compiled, not run on hardware
//
#include <hip/hip_runtime.h>
#include <hip/hip_bf16.h>
#include <stdint.h>

typedef __attribute__((ext_vector_type(16))) __bf16 v16bf;
typedef __attribute__((ext_vector_type(8)))  __bf16 v8bf;
typedef __attribute__((ext_vector_type(8)))  float  v8f;

#define NCODES 8192
#define DIM    512
#define NVEC   32768
#define EMA    0.99f
#define ONE_M_EMA 0.01f
#define EPSV   1e-5f

#define CAT16(lo,hi) __builtin_shufflevector((lo),(hi),0,1,2,3,4,5,6,7,8,9,10,11,12,13,14,15)

// ---------------- prep: convert E to bf16, b2 = ||E_k||^2, zero accumulators ----------------
__global__ __launch_bounds__(256)
void prep_codes(const float* __restrict__ E, __bf16* __restrict__ Eb,
                float* __restrict__ b2, float* __restrict__ counts,
                float* __restrict__ embed_sum, float* __restrict__ n_total) {
    int k = blockIdx.x, t = threadIdx.x;
    int lane = t & 31, wave = t >> 5;
    float s = 0.0f;
#pragma unroll
    for (int i = 0; i < 2; ++i) {
        int d = t + i * 256;
        float v = E[(size_t)k * DIM + d];
        Eb[(size_t)k * DIM + d] = (__bf16)v;
        embed_sum[(size_t)k * DIM + d] = 0.0f;
        s += v * v;
    }
#pragma unroll
    for (int off = 16; off > 0; off >>= 1) s += __shfl_down(s, off, 32);
    __shared__ float red[8];
    if (lane == 0) red[wave] = s;
    __syncthreads();
    if (t == 0) {
        float tot = 0.0f;
#pragma unroll
        for (int i = 0; i < 8; ++i) tot += red[i];
        b2[k] = tot;
        counts[k] = 0.0f;
        if (blockIdx.x == 0) *n_total = 0.0f;
    }
}

// ---------------- prep: convert z to bf16 ----------------
__global__ __launch_bounds__(256)
void prep_z(const float* __restrict__ z, __bf16* __restrict__ zb) {
    size_t base = ((size_t)blockIdx.x * 256 + threadIdx.x) * 8;
#pragma unroll
    for (int j = 0; j < 8; ++j) zb[base + j] = (__bf16)z[base + j];
}

// ---------------- GEMM (bf16 WMMA) + row argmin of  b2[k] - 2 * z.E_k ----------------
// block = 256 threads (8 waves), 128 rows per block; each wave owns 16 rows.
#define LDS_ROW 520   // padded row (bf16 elems): 520*2B=1040B -> bank stride 4, conflict-free

__global__ __launch_bounds__(256)
void gemm_argmin(const __bf16* __restrict__ zb, const __bf16* __restrict__ Eb,
                 const float* __restrict__ b2, int* __restrict__ idx_out,
                 float* __restrict__ idx_f_out) {
    __shared__ __bf16 lds[2][16 * LDS_ROW];
    const int t = threadIdx.x;
    const int wave = t >> 5, lane = t & 31;
    const int lrow = lane & 15, lhalf = lane >> 4;
    const int rowbase = blockIdx.x * 128 + wave * 16;

    // ---- load this wave's A fragments (16 rows x 512) into registers, ISA bf16 A layout ----
    v16bf a[16];
    {
        const __bf16* zrow = zb + (size_t)(rowbase + lrow) * DIM;
        const int koff = lhalf * 8;
#pragma unroll
        for (int j = 0; j < 16; ++j) {
            v8bf lo = *(const v8bf*)(zrow + 32 * j + koff);
            v8bf hi = *(const v8bf*)(zrow + 32 * j + 16 + koff);
            a[j] = CAT16(lo, hi);
        }
    }

    float runmin[8];
    int   runidx[8];
#pragma unroll
    for (int r = 0; r < 8; ++r) { runmin[r] = 3.402823466e38f; runidx[r] = 0; }

    // ---- stage first 16-code tile of E into LDS ----
    uint4 stage[4];
#pragma unroll
    for (int i = 0; i < 4; ++i) {
        int q = t + i * 256;                 // uint4 slot in 16x512 bf16 tile
        int row = q >> 6, c16 = q & 63;
        stage[i] = *(const uint4*)(Eb + (size_t)row * DIM + c16 * 8);
    }
#pragma unroll
    for (int i = 0; i < 4; ++i) {
        int q = t + i * 256;
        int row = q >> 6, c16 = q & 63;
        *(uint4*)(&lds[0][row * LDS_ROW + c16 * 8]) = stage[i];
    }

    const int NCHUNK = NCODES / 16;          // 512 column chunks
    for (int c = 0; c < NCHUNK; ++c) {
        __syncthreads();
        if (c + 1 < NCHUNK) {
            const int cb = (c + 1) * 16;
#pragma unroll
            for (int i = 0; i < 4; ++i) {
                int q = t + i * 256;
                int row = q >> 6, c16 = q & 63;
                stage[i] = *(const uint4*)(Eb + (size_t)(cb + row) * DIM + c16 * 8);
            }
        }
        if (c + 2 < NCHUNK)
            __builtin_prefetch(Eb + (size_t)(c + 2) * 16 * DIM, 0, 3);

        v8f acc = {};
        const __bf16* brow = &lds[c & 1][lrow * LDS_ROW + lhalf * 8];
#pragma unroll
        for (int j = 0; j < 16; ++j) {
            v8bf b0 = *(const v8bf*)(brow + 32 * j);
            v8bf b1 = *(const v8bf*)(brow + 32 * j + 16);
            v16bf bfrag = CAT16(b0, b1);
            acc = __builtin_amdgcn_wmma_f32_16x16x32_bf16(
                false, a[j], false, bfrag, (short)0, acc, false, false);
        }

        if (c + 1 < NCHUNK) {
#pragma unroll
            for (int i = 0; i < 4; ++i) {
                int q = t + i * 256;
                int row = q >> 6, c16 = q & 63;
                *(uint4*)(&lds[(c + 1) & 1][row * LDS_ROW + c16 * 8]) = stage[i];
            }
        }

        const int col = c * 16 + lrow;       // this lane's column (C layout: N = lane%16)
        const float bb = b2[col];
#pragma unroll
        for (int r = 0; r < 8; ++r) {        // rows M = lhalf*8 + r
            float dist = bb - 2.0f * acc[r];
            if (dist < runmin[r]) { runmin[r] = dist; runidx[r] = col; }
        }
    }

    // ---- reduce over the 16 lanes of each half-wave (butterfly stays within half) ----
#pragma unroll
    for (int off = 1; off < 16; off <<= 1) {
#pragma unroll
        for (int r = 0; r < 8; ++r) {
            float om = __shfl_xor(runmin[r], off, 32);
            int   oi = __shfl_xor(runidx[r], off, 32);
            if (om < runmin[r] || (om == runmin[r] && oi < runidx[r])) {
                runmin[r] = om; runidx[r] = oi;
            }
        }
    }
    if (lrow == 0) {
#pragma unroll
        for (int r = 0; r < 8; ++r) {
            int row = rowbase + lhalf * 8 + r;
            idx_out[row]   = runidx[r];
            idx_f_out[row] = (float)runidx[r];
        }
    }
}

// ---------------- gather quantized rows + scatter counts / embed_sum ----------------
__global__ __launch_bounds__(256)
void gather_scatter(const float* __restrict__ z, const float* __restrict__ E,
                    const int* __restrict__ idx, float* __restrict__ out_q,
                    float* __restrict__ counts, float* __restrict__ embed_sum) {
    int n = blockIdx.x, t = threadIdx.x;
    int k = idx[n];
#pragma unroll
    for (int i = 0; i < 2; ++i) {
        int d = t + i * 256;
        out_q[(size_t)n * DIM + d] = E[(size_t)k * DIM + d];
        atomicAdd(&embed_sum[(size_t)k * DIM + d], z[(size_t)n * DIM + d]);
    }
    if (t == 0) atomicAdd(&counts[k], 1.0f);
}

// ---------------- EMA cluster sizes + global sum n ----------------
__global__ __launch_bounds__(256)
void ema_cs(const float* __restrict__ cluster_size, const float* __restrict__ counts,
            float* __restrict__ cs_ws, float* __restrict__ n_total) {
    int k = blockIdx.x * 256 + threadIdx.x;
    int lane = threadIdx.x & 31, wave = threadIdx.x >> 5;
    float cs = cluster_size[k] * EMA + ONE_M_EMA * counts[k];
    cs_ws[k] = cs;
    float s = cs;
#pragma unroll
    for (int off = 16; off > 0; off >>= 1) s += __shfl_down(s, off, 32);
    __shared__ float red[8];
    if (lane == 0) red[wave] = s;
    __syncthreads();
    if (threadIdx.x == 0) {
        float tot = 0.0f;
#pragma unroll
        for (int i = 0; i < 8; ++i) tot += red[i];
        atomicAdd(n_total, tot);
    }
}

// ---------------- prefix scan of dead flags -> replacement ranks ----------------
__global__ __launch_bounds__(1024)
void dead_scan(const float* __restrict__ cs_ws, int* __restrict__ ranks) {
    __shared__ int s[1024];
    int t = threadIdx.x;
    int base = t * 8, loc[8], sum = 0;
#pragma unroll
    for (int i = 0; i < 8; ++i) {
        sum += (cs_ws[base + i] < 1.0f) ? 1 : 0;
        loc[i] = sum;                        // inclusive within this thread's span
    }
    s[t] = sum;
    __syncthreads();
    for (int off = 1; off < 1024; off <<= 1) {
        int v = (t >= off) ? s[t - off] : 0;
        __syncthreads();
        s[t] += v;
        __syncthreads();
    }
    int pre = (t > 0) ? s[t - 1] : 0;
#pragma unroll
    for (int i = 0; i < 8; ++i) {
        int r = pre + loc[i] - 1;
        if (r < 0) r = 0;
        if (r > NVEC - 1) r = NVEC - 1;
        ranks[base + i] = r;
    }
}

// ---------------- finalize emb / cs / ea with dead-code reinit ----------------
__global__ __launch_bounds__(128)
void finalize(const float* __restrict__ z, const float* __restrict__ embed_avg,
              const float* __restrict__ embed_sum, const float* __restrict__ cs_ws,
              const int* __restrict__ ranks, const int* __restrict__ perm,
              const float* __restrict__ n_total,
              float* __restrict__ out_emb, float* __restrict__ out_cs,
              float* __restrict__ out_ea) {
    int k = blockIdx.x, t = threadIdx.x;
    float cs = cs_ws[k];
    bool dead = cs < 1.0f;
    float n = *n_total;
    float smoothed = (cs + EPSV) / (n + (float)NCODES * EPSV) * n;
    if (dead) {
        int src = perm[ranks[k]];
#pragma unroll
        for (int i = 0; i < 4; ++i) {
            int d = t + i * 128;
            float r = z[(size_t)src * DIM + d];
            out_emb[(size_t)k * DIM + d] = r;
            out_ea[(size_t)k * DIM + d]  = r;
        }
        if (t == 0) out_cs[k] = 1.0f;
    } else {
        float inv = 1.0f / smoothed;
#pragma unroll
        for (int i = 0; i < 4; ++i) {
            int d = t + i * 128;
            float ea = embed_avg[(size_t)k * DIM + d] * EMA +
                       ONE_M_EMA * embed_sum[(size_t)k * DIM + d];
            out_ea[(size_t)k * DIM + d]  = ea;
            out_emb[(size_t)k * DIM + d] = ea * inv;
        }
        if (t == 0) out_cs[k] = cs;
    }
}

extern "C" void kernel_launch(void* const* d_in, const int* in_sizes, int n_in,
                              void* d_out, int out_size, void* d_ws, size_t ws_size,
                              hipStream_t stream) {
    const float* z            = (const float*)d_in[0];   // [N, D]
    const float* embedding    = (const float*)d_in[1];   // [K, D]
    const float* cluster_size = (const float*)d_in[2];   // [K]
    const float* embed_avg    = (const float*)d_in[3];   // [K, D]
    const int*   perm         = (const int*)d_in[4];     // [N]

    float* out = (float*)d_out;
    float* out_q   = out;                                   // [N,D]
    float* out_idx = out + (size_t)NVEC * DIM;              // [N]
    float* out_emb = out_idx + NVEC;                        // [K,D]
    float* out_cs  = out_emb + (size_t)NCODES * DIM;        // [K]
    float* out_ea  = out_cs + NCODES;                       // [K,D]

    char* ws = (char*)d_ws;
    __bf16* Eb        = (__bf16*)(ws);                      // 8 MB
    __bf16* zb        = (__bf16*)(ws + 8388608);            // 32 MB
    float*  b2        = (float*) (ws + 41943040);
    float*  counts    = (float*) (ws + 41975808);
    float*  cs_ws     = (float*) (ws + 42008576);
    int*    ranks     = (int*)   (ws + 42041344);
    int*    idxbuf    = (int*)   (ws + 42074112);
    float*  n_total   = (float*) (ws + 42205184);
    float*  embed_sum = (float*) (ws + 42205440);           // 16 MB

    prep_codes<<<NCODES, 256, 0, stream>>>(embedding, Eb, b2, counts, embed_sum, n_total);
    prep_z<<<NVEC * DIM / (256 * 8), 256, 0, stream>>>(z, zb);
    gemm_argmin<<<NVEC / 128, 256, 0, stream>>>(zb, Eb, b2, idxbuf, out_idx);
    gather_scatter<<<NVEC, 256, 0, stream>>>(z, embedding, idxbuf, out_q, counts, embed_sum);
    ema_cs<<<NCODES / 256, 256, 0, stream>>>(cluster_size, counts, cs_ws, n_total);
    dead_scan<<<1, 1024, 0, stream>>>(cs_ws, ranks);
    finalize<<<NCODES, 128, 0, stream>>>(z, embed_avg, embed_sum, cs_ws, ranks, perm,
                                         n_total, out_emb, out_cs, out_ea);
}
